// SimplifiedMultiHeadSelfAttention_86303072845998
// MI455X (gfx1250) — matrix operand order repaired and checked by reading, not compiled, not run on hardware
//
#include <hip/hip_runtime.h>

// ---------------------------------------------------------------------------
// Fused MHSA for MI455X (gfx1250): f16 WMMA (16x16x32) everywhere.
//   B=2, T=2048, D=768, H=12, DH=64
// v3: Tensor Data Mover (TENSOR_LOAD_TO_LDS) drives the double-buffered K/V
//     staging in attention (TENSORcnt-tracked DMA, zero VALU copy cost);
//     pack_w becomes a coalesced LDS tile transpose.
// ---------------------------------------------------------------------------

typedef _Float16 h8  __attribute__((ext_vector_type(8)));
typedef _Float16 h16 __attribute__((ext_vector_type(16)));
typedef float    f8  __attribute__((ext_vector_type(8)));
typedef unsigned int u32;
typedef u32  v4u __attribute__((ext_vector_type(4)));
typedef int  v4i __attribute__((ext_vector_type(4)));
typedef int  v8i __attribute__((ext_vector_type(8)));

#define TT   2048
#define DD   768
#define HH   12
#define DHD  64
#define BB   2
#define BHN  (BB*HH)        // 24
#define ROWS (BB*TT)        // 4096
#define NCOL (3*DD)         // 2304
#define NPAIR (TT/32)       // 64 key-pair iterations
// fold DH^-0.5 * log2(e) into Q so softmax uses exp2 (v_exp_f32)
#define QSCALE 0.1803368801111204f

static __device__ __forceinline__ h16 cat16(h8 lo, h8 hi) {
  return __builtin_shufflevector(lo, hi, 0,1,2,3,4,5,6,7,8,9,10,11,12,13,14,15);
}
static __device__ __forceinline__ h16 ld16(const _Float16* p) {
  return cat16(*(const h8*)p, *(const h8*)(p + 8));
}
static __device__ __forceinline__ float rmax16(float v) {
  v = fmaxf(v, __shfl_xor(v, 1, 16));
  v = fmaxf(v, __shfl_xor(v, 2, 16));
  v = fmaxf(v, __shfl_xor(v, 4, 16));
  v = fmaxf(v, __shfl_xor(v, 8, 16));
  return v;
}
static __device__ __forceinline__ float rsum16(float v) {
  v += __shfl_xor(v, 1, 16);
  v += __shfl_xor(v, 2, 16);
  v += __shfl_xor(v, 4, 16);
  v += __shfl_xor(v, 8, 16);
  return v;
}
static __device__ __forceinline__ float fexp2(float x) {
  return __builtin_amdgcn_exp2f(x);
}
static __device__ __forceinline__ f8 wmma16(h16 a, h16 b, f8 c) {
  return __builtin_amdgcn_wmma_f32_16x16x32_f16(false, a, false, b, (short)0, c,
                                                false, false);
}

// --------- Tensor Data Mover: 2D tile (global -> LDS), dims in DWORDs -------
// D# per CDNA5 ISA ch.8: group0 = {count|flags, lds_addr, global_addr, type=2},
// group1 = {data_size=4B, tensor_dim0/1, tile_dim0/1, dim0_stride}.
static __device__ __forceinline__ void tdm_load_2d(u32 lds_off, const void* gptr,
                                                   u32 tensor_w, u32 tensor_h,
                                                   u32 row_stride, u32 tile_w,
                                                   u32 tile_h) {
  unsigned long long ga = (unsigned long long)gptr;
  v4u g0;
  g0[0] = 1u;                                            // count=1, no gather
  g0[1] = lds_off;                                       // LDS byte address
  g0[2] = (u32)ga;                                       // global_addr[31:0]
  g0[3] = (u32)((ga >> 32) & 0x1FFFFFFu) | (2u << 30);   // addr[56:32]|type=2
  v8i g1;
  g1[0] = (int)(2u << 16);                               // data_size=4B
  g1[1] = (int)(tensor_w << 16);                         // tensor_dim0 lo
  g1[2] = (int)((tensor_w >> 16) | (tensor_h << 16));    // dim0 hi | dim1 lo
  g1[3] = (int)((tensor_h >> 16) | (tile_w << 16));      // dim1 hi | tile_dim0
  g1[4] = (int)(tile_h & 0xFFFFu);                       // tile_dim1 (dim2=0)
  g1[5] = (int)row_stride;                               // dim0_stride lo
  g1[6] = 0;                                             // stride hi / dim1_stride
  g1[7] = 0;
  v4i z = {0, 0, 0, 0};
#if defined(__clang_major__) && __clang_major__ >= 23
  v8i z8 = {0, 0, 0, 0, 0, 0, 0, 0};
  __builtin_amdgcn_tensor_load_to_lds(g0, g1, z, z, z8, 0);
#else
  __builtin_amdgcn_tensor_load_to_lds(g0, g1, z, z, 0);
#endif
}

// ------------------------- pack kernels ------------------------------------
__global__ void pack_x_kernel(const float* __restrict__ x, _Float16* __restrict__ xh, int n) {
  int i = blockIdx.x * 256 + threadIdx.x;
  if (i < n) xh[i] = (_Float16)x[i];
}
// W [D, 3D] fp32 -> Wt [3D, D] f16 via 32x32 LDS tile transpose (coalesced
// fp32 reads AND coalesced f16 writes).
__global__ void __launch_bounds__(256)
pack_w_kernel(const float* __restrict__ w, _Float16* __restrict__ wt) {
  __shared__ _Float16 tile[32][33];
  const int tx = threadIdx.x & 31;
  const int ty = threadIdx.x >> 5;                 // 0..7
  const int k0 = (blockIdx.x % (DD / 32)) * 32;    // 24 k-tiles
  const int c0 = (blockIdx.x / (DD / 32)) * 32;    // 72 c-tiles
#pragma unroll
  for (int i = 0; i < 4; ++i)
    tile[ty + i * 8][tx] = (_Float16)w[(size_t)(k0 + ty + i * 8) * NCOL + c0 + tx];
  __syncthreads();
#pragma unroll
  for (int i = 0; i < 4; ++i)
    wt[(size_t)(c0 + ty + i * 8) * DD + k0 + tx] = tile[tx][ty + i * 8];
}

// ------------------------- QKV projection GEMM -----------------------------
// One 32x64 strip per wave (2 M-tiles x 4 N-tiles): 8 WMMA per K-step with
// 4x A-frag and 2x B-frag register reuse.
__global__ void __launch_bounds__(128)
qkv_gemm_kernel(const _Float16* __restrict__ xh, const _Float16* __restrict__ wt,
                const float* __restrict__ bias,
                _Float16* __restrict__ Qh, _Float16* __restrict__ Kh,
                _Float16* __restrict__ Vt) {
  const int lane = threadIdx.x & 31, g = lane >> 4, n16 = lane & 15;
  const int tid  = blockIdx.x * 4 + (threadIdx.x >> 5);
  const int mt   = tid / (NCOL / 64);   // 0..127 : 32-row strip
  const int nt   = tid % (NCOL / 64);   // 0..35  : 64-col strip

  const _Float16* a0 = xh + (size_t)(mt * 32 + n16) * DD;       // A: M = lane%16
  const _Float16* a1 = a0 + (size_t)16 * DD;
  const _Float16* b0 = wt + (size_t)(nt * 64 + n16) * DD;       // B: N = lane%16

  f8 acc[2][4] = {};
#pragma unroll 2
  for (int kt = 0; kt < DD / 32; ++kt) {
    h16 A0 = cat16(*(const h8*)(a0 + kt * 32 + 8 * g),
                   *(const h8*)(a0 + kt * 32 + 16 + 8 * g));
    h16 A1 = cat16(*(const h8*)(a1 + kt * 32 + 8 * g),
                   *(const h8*)(a1 + kt * 32 + 16 + 8 * g));
#pragma unroll
    for (int c = 0; c < 4; ++c) {
      h16 Bc = cat16(*(const h8*)(b0 + (size_t)(16 * c) * DD + kt * 32 + 8 * g),
                     *(const h8*)(b0 + (size_t)(16 * c) * DD + kt * 32 + 16 + 8 * g));
      acc[0][c] = wmma16(A0, Bc, acc[0][c]);
      acc[1][c] = wmma16(A1, Bc, acc[1][c]);
    }
  }

  const int which = (nt * 64) / DD;            // 0=q 1=k 2=v (uniform per wave)
  const int hidx  = ((nt * 64) % DD) / DHD;    // head (uniform: strip = one head)
  const int b     = (mt * 32) >> 11;
  const int t0    = (mt * 32) & (TT - 1);
  const int bh    = b * HH + hidx;

  float bv[4];
#pragma unroll
  for (int c = 0; c < 4; ++c) bv[c] = bias[nt * 64 + c * 16 + n16];

  if (which == 0) {
#pragma unroll
    for (int mi = 0; mi < 2; ++mi)
#pragma unroll
      for (int c = 0; c < 4; ++c) {
        _Float16* q = Qh + ((size_t)bh * TT) * DHD + c * 16 + n16;
#pragma unroll
        for (int r = 0; r < 8; ++r)
          q[(size_t)(t0 + mi * 16 + r + 8 * g) * DHD] =
              (_Float16)((acc[mi][c][r] + bv[c]) * QSCALE);
      }
  } else if (which == 1) {
#pragma unroll
    for (int mi = 0; mi < 2; ++mi)
#pragma unroll
      for (int c = 0; c < 4; ++c) {
        _Float16* k = Kh + ((size_t)bh * TT) * DHD + c * 16 + n16;
#pragma unroll
        for (int r = 0; r < 8; ++r)
          k[(size_t)(t0 + mi * 16 + r + 8 * g) * DHD] =
              (_Float16)(acc[mi][c][r] + bv[c]);
      }
  } else {
#pragma unroll
    for (int mi = 0; mi < 2; ++mi)
#pragma unroll
      for (int c = 0; c < 4; ++c) {
        h8 v;
#pragma unroll
        for (int r = 0; r < 8; ++r) v[r] = (_Float16)(acc[mi][c][r] + bv[c]);
        *(h8*)(Vt + ((size_t)(bh * DHD + c * 16 + n16)) * TT + t0 + mi * 16 + 8 * g) = v;
      }
  }
}

// ------------------------- flash attention ---------------------------------
// Block = 4 waves, 64 query rows. K/V tiles (32 keys) DMA'd into LDS by the
// Tensor Data Mover (wave 0 issues the descriptors; TENSORcnt tracks them),
// double-buffered so the DMA of tile p+1 overlaps compute on tile p.
__global__ void __launch_bounds__(128)
attn_kernel(const _Float16* __restrict__ Qh, const _Float16* __restrict__ Kh,
            const _Float16* __restrict__ Vt, float* __restrict__ out) {
  __shared__ __align__(16) _Float16 Ks[2][32][64];   // 2 x 4KB  (keys x dh)
  __shared__ __align__(16) _Float16 Vs[2][64][32];   // 2 x 4KB  (dh x keys)
  __shared__ __align__(16) _Float16 Ps[4][16][32];   // 4KB      (per-wave P)

  const int tidb = threadIdx.x;                       // 0..127
  const int lane = tidb & 31, g = lane >> 4, n16 = lane & 15;
  const int wv   = tidb >> 5;
  const int bh   = blockIdx.x / (TT / 64);
  const int qblk = blockIdx.x % (TT / 64);
  const int qrow0 = qblk * 64 + wv * 16;
  const int b = bh / HH, hd = bh % HH;

  const char* kbase_g = (const char*)(Kh + (size_t)bh * TT * DHD);   // rows: 128B
  const char* vbase_g = (const char*)(Vt + (size_t)bh * DHD * TT);   // rows: 4KB
  const u32 ks_off[2] = { (u32)(size_t)&Ks[0][0][0], (u32)(size_t)&Ks[1][0][0] };
  const u32 vs_off[2] = { (u32)(size_t)&Vs[0][0][0], (u32)(size_t)&Vs[1][0][0] };

  // Q A-frags (held in VGPRs for the whole pass); scale pre-folded.
  const _Float16* qrow = Qh + ((size_t)bh * TT + qrow0 + n16) * DHD;
  h16 qa0 = cat16(*(const h8*)(qrow + 8 * g),      *(const h8*)(qrow + 16 + 8 * g));
  h16 qa1 = cat16(*(const h8*)(qrow + 32 + 8 * g), *(const h8*)(qrow + 48 + 8 * g));

  f8 acc0 = {}, acc1 = {}, acc2 = {}, acc3 = {};
  float m[8], l[8];
#pragma unroll
  for (int r = 0; r < 8; ++r) { m[r] = -3.0e38f; l[r] = 0.0f; }

  // prime buffer 0 with tile p=0 (K: 32 rows x 32 dw; V: 64 rows x 16 dw)
  if (wv == 0) {
    tdm_load_2d(ks_off[0], kbase_g, 32, TT, 32, 32, 32);
    tdm_load_2d(vs_off[0], vbase_g, TT / 2, DHD, TT / 2, 16, DHD);
  }

  for (int p = 0; p < NPAIR; ++p) {
    if (wv == 0) __builtin_amdgcn_s_wait_tensorcnt(0);  // fill(p) landed in LDS
    __syncthreads();   // publish buf[p&1]; all reads of buf[(p+1)&1] retired

    // ---- TDM: stage tile p+1 into the other buffer (overlaps compute) ----
    if (wv == 0 && p + 1 < NPAIR) {
      const int nxt = (p + 1) & 1;
      tdm_load_2d(ks_off[nxt], kbase_g + (size_t)(p + 1) * 4096, 32, TT, 32, 32, 32);
      tdm_load_2d(vs_off[nxt], vbase_g + (size_t)(p + 1) * 64, TT / 2, DHD, TT / 2, 16, DHD);
    }
    if (p + 2 < NPAIR) {  // warm L2 for the DMA after next (global_prefetch_b8)
      __builtin_prefetch(kbase_g + (size_t)(p + 2) * 4096 + tidb * 32, 0, 0);
      __builtin_prefetch(vbase_g + (size_t)(tidb & 63) * (TT * 2) + (p + 2) * 64, 0, 0);
    }

    const int cur = p & 1;
    // ---- scores for key tiles j0=2p, j1=2p+1 (K frags from LDS) ----
    h16 kb00 = cat16(*(const h8*)&Ks[cur][n16][8 * g], *(const h8*)&Ks[cur][n16][16 + 8 * g]);
    h16 kb01 = cat16(*(const h8*)&Ks[cur][n16][32 + 8 * g], *(const h8*)&Ks[cur][n16][48 + 8 * g]);
    h16 kb10 = cat16(*(const h8*)&Ks[cur][16 + n16][8 * g], *(const h8*)&Ks[cur][16 + n16][16 + 8 * g]);
    h16 kb11 = cat16(*(const h8*)&Ks[cur][16 + n16][32 + 8 * g], *(const h8*)&Ks[cur][16 + n16][48 + 8 * g]);
    f8 s0 = {}, s1 = {};
    s0 = wmma16(qa0, kb00, s0);
    s0 = wmma16(qa1, kb01, s0);
    s1 = wmma16(qa0, kb10, s1);
    s1 = wmma16(qa1, kb11, s1);

    // ---- online softmax (base-2; row M = r + 8g lives across 16 lanes) ----
#pragma unroll
    for (int r = 0; r < 8; ++r) {
      float rm = rmax16(fmaxf(s0[r], s1[r]));
      float mn = fmaxf(m[r], rm);
      float alpha = fexp2(m[r] - mn);
      float p0 = fexp2(s0[r] - mn);
      float p1 = fexp2(s1[r] - mn);
      Ps[wv][r + 8 * g][n16]      = (_Float16)p0;
      Ps[wv][r + 8 * g][16 + n16] = (_Float16)p1;
      l[r] = l[r] * alpha + rsum16(p0 + p1);
      m[r] = mn;
      acc0[r] *= alpha; acc1[r] *= alpha; acc2[r] *= alpha; acc3[r] *= alpha;
    }

    // ---- P (C-layout) -> A-layout via LDS, then PV (V frags from LDS) ----
    const _Float16* prow = &Ps[wv][n16][0];
    h16 pa = cat16(*(const h8*)(prow + 8 * g), *(const h8*)(prow + 16 + 8 * g));

    h16 vb0 = cat16(*(const h8*)&Vs[cur][0 * 16 + n16][8 * g], *(const h8*)&Vs[cur][0 * 16 + n16][16 + 8 * g]);
    h16 vb1 = cat16(*(const h8*)&Vs[cur][1 * 16 + n16][8 * g], *(const h8*)&Vs[cur][1 * 16 + n16][16 + 8 * g]);
    h16 vb2 = cat16(*(const h8*)&Vs[cur][2 * 16 + n16][8 * g], *(const h8*)&Vs[cur][2 * 16 + n16][16 + 8 * g]);
    h16 vb3 = cat16(*(const h8*)&Vs[cur][3 * 16 + n16][8 * g], *(const h8*)&Vs[cur][3 * 16 + n16][16 + 8 * g]);
    acc0 = wmma16(pa, vb0, acc0);
    acc1 = wmma16(pa, vb1, acc1);
    acc2 = wmma16(pa, vb2, acc2);
    acc3 = wmma16(pa, vb3, acc3);
  }

  // ---- epilogue: O /= l, write fp32 out[b, t, hd*64 + dh] ----
  float* orow = out + ((size_t)b * TT + qrow0) * DD + hd * DHD + n16;
#pragma unroll
  for (int r = 0; r < 8; ++r) {
    float inv = __builtin_amdgcn_rcpf(l[r]);
    float* o = orow + (size_t)(r + 8 * g) * DD;
    o[0]  = acc0[r] * inv;
    o[16] = acc1[r] * inv;
    o[32] = acc2[r] * inv;
    o[48] = acc3[r] * inv;
  }
}

// ---------------------------------------------------------------------------
extern "C" void kernel_launch(void* const* d_in, const int* in_sizes, int n_in,
                              void* d_out, int out_size, void* d_ws, size_t ws_size,
                              hipStream_t stream) {
  const float* x    = (const float*)d_in[0];
  const float* Wqkv = (const float*)d_in[1];
  const float* bqkv = (const float*)d_in[2];
  float* out        = (float*)d_out;

  char* ws = (char*)d_ws;
  _Float16* xh = (_Float16*)(ws);                               // 4096*768
  _Float16* wt = (_Float16*)(ws + (size_t)ROWS * DD * 2);       // 2304*768
  _Float16* qh = (_Float16*)(ws + (size_t)ROWS * DD * 2 + (size_t)DD * NCOL * 2);
  _Float16* kh = qh + (size_t)BHN * TT * DHD;
  _Float16* vt = kh + (size_t)BHN * TT * DHD;

  const int nx = ROWS * DD;          // 3,145,728
  pack_x_kernel<<<(nx + 255) / 256, 256, 0, stream>>>(x, xh, nx);
  // 24 k-tiles x 72 c-tiles of 32x32
  pack_w_kernel<<<(DD / 32) * (NCOL / 32), 256, 0, stream>>>(Wqkv, wt);

  // 128 x 36 strips (32x64 each), 4 waves / block
  qkv_gemm_kernel<<<(128 * 36) / 4, 128, 0, stream>>>(xh, wt, bqkv, qh, kh, vt);

  // 24 heads x 32 query blocks (64 rows each, 4 waves / block)
  attn_kernel<<<BHN * (TT / 64), 128, 0, stream>>>(qh, kh, vt, out);
}
